// PureSSNLayer_34651796144534
// MI455X (gfx1250) — matrix operand order repaired and checked by reading, not compiled, not run on hardware
//
#include <hip/hip_runtime.h>
#include <hip/hip_bf16.h>
#include <math.h>

typedef __attribute__((ext_vector_type(16))) _Float16 v16h;
typedef __attribute__((ext_vector_type(8)))  float    v8f;
typedef __attribute__((ext_vector_type(4)))  unsigned int u32x4;
typedef __attribute__((ext_vector_type(8)))  int          i32x8;
typedef __attribute__((ext_vector_type(4)))  int          i32x4;

#define DM      768
#define BATCH   16
#define TLEN    1024
#define DPAD    800      // padded total state dim (E 464 | PV 144 | SOM 112 | VIP 80)
#define NTILES  50       // DPAD / 16
#define KCH     25       // DPAD / 32
#define KCH_IN  24       // DM / 32
#define NT_OUT  48       // DM / 16

// real sizes
#define D_E   460
#define D_PV  138
#define D_S   107
#define D_V    63
// padded region bases
#define P_E   0
#define P_PV  464
#define P_S   608
#define P_V   720

#if defined(__AMDGCN__) && __has_builtin(__builtin_amdgcn_tensor_load_to_lds) && \
    __has_builtin(__builtin_amdgcn_s_wait_tensorcnt)
#define HAVE_TDM 1
#endif

__device__ __forceinline__ float softplusf(float x) {
    return (x > 20.f) ? x : log1pf(expf(x));
}

// padded dim -> packed (concat yE|yPV|yS|yV) index, -1 for pad lanes
__device__ __forceinline__ int packed_index(int n) {
    if (n < 460) return n;
    if (n < 464) return -1;
    if (n < 602) return n - 4;       // PV: 460 + (n-464)
    if (n < 608) return -1;
    if (n < 715) return n - 10;      // SOM: 598 + (n-608)
    if (n < 720) return -1;
    if (n < 783) return n - 15;      // VIP: 705 + (n-720)
    return -1;
}

// ---------------- fragment helpers (contiguous k order, wave32) ----------------
// A 16x32 f16 fragment: lane l -> row M=l&15; elems 0..15 = K 2*kb..2*kb+15, kb=(l>>4)*8
__device__ __forceinline__ v16h load_a_row(const _Float16* rowp, int kb) {
    v16h a;
#pragma unroll
    for (int i = 0; i < 8; ++i) {
        a[i]     = rowp[2 * kb + i];
        a[i + 8] = rowp[2 * kb + 8 + i];
    }
    return a;
}
// B 32x16 f16 fragment, pre-swizzled: 32 lanes x 16 contiguous halves (32B/lane)
__device__ __forceinline__ v16h load_b_swz(const _Float16* frag) {
    const v16h* p = (const v16h*)frag;
    return p[threadIdx.x & 31];
}

// ---------------- TDM: 1-D contiguous global -> LDS DMA (data_size = 4B) ----------------
#ifdef HAVE_TDM
__device__ __forceinline__ void tdm_load_1d(unsigned lds_off, const void* gptr, unsigned nbytes) {
    unsigned long long ga = (unsigned long long)gptr;
    unsigned n32 = nbytes >> 2;                          // dwords, fits tile_dim0 (16b)
    u32x4 g0;
    g0[0] = 1u;                                          // count=1 (valid user descriptor)
    g0[1] = lds_off;                                     // lds_addr (bytes)
    g0[2] = (unsigned)(ga & 0xFFFFFFFFu);                // global_addr[31:0]
    g0[3] = (unsigned)((ga >> 32) & 0x01FFFFFFu) | (2u << 30);  // addr[56:32] | type=2
    i32x8 g1;
    g1[0] = (int)(2u << 16);                             // data_size=2 -> 4 bytes
    g1[1] = (int)((n32 & 0xFFFFu) << 16);                // tensor_dim0[15:0]
    g1[2] = (int)(((n32 >> 16) & 0xFFFFu) | (1u << 16)); // tensor_dim0[31:16], tensor_dim1=1
    g1[3] = (int)((n32 & 0xFFFFu) << 16);                // tile_dim0 = n32
    g1[4] = 1;                                           // tile_dim1 = 1
    g1[5] = (int)n32;                                    // tensor_dim0_stride[31:0]
    g1[6] = 0;
    g1[7] = 0;
    i32x4 z4 = {0, 0, 0, 0};
#if defined(__clang_major__) && __clang_major__ >= 23
    i32x8 z8 = {0, 0, 0, 0, 0, 0, 0, 0};
    __builtin_amdgcn_tensor_load_to_lds(g0, g1, z4, z4, z8, 0);
#else
    __builtin_amdgcn_tensor_load_to_lds(g0, g1, z4, z4, 0);
#endif
}
#endif

// ---------------- prep: h -> f16 ----------------
__global__ __launch_bounds__(256) void h_to_f16_kernel(const float* __restrict__ h,
                                                       _Float16* __restrict__ h16, int n) {
    int i = blockIdx.x * 256 + threadIdx.x;
    if (i < n) h16[i] = (_Float16)h[i];
}

// ---------------- prep: fused Dale weight matrix, pre-swizzled ----------------
__device__ float wbig_val(int n, int k,
                          const float* EE, const float* PVE, const float* SE, const float* VE,
                          const float* EPV, const float* PVPV, const float* SPV,
                          const float* ES, const float* PVS, const float* VS, const float* SV) {
    int ro, lo;
    if (n < P_PV)      { ro = 0; lo = n;        if (lo >= D_E)  return 0.f; }
    else if (n < P_S)  { ro = 1; lo = n - P_PV; if (lo >= D_PV) return 0.f; }
    else if (n < P_V)  { ro = 2; lo = n - P_S;  if (lo >= D_S)  return 0.f; }
    else               { ro = 3; lo = n - P_V;  if (lo >= D_V)  return 0.f; }
    int ri, li, din;
    if (k < P_PV)      { ri = 0; li = k;        if (li >= D_E)  return 0.f; din = D_E; }
    else if (k < P_S)  { ri = 1; li = k - P_PV; if (li >= D_PV) return 0.f; din = D_PV; }
    else if (k < P_V)  { ri = 2; li = k - P_S;  if (li >= D_S)  return 0.f; din = D_S; }
    else               { ri = 3; li = k - P_V;  if (li >= D_V)  return 0.f; din = D_V; }
    const float* w = nullptr; float s = 0.f;
    switch (ro * 4 + ri) {
        case 0:  w = EE;   s =  1.f; break;   // iE <- yE
        case 1:  w = EPV;  s = -1.f; break;   // iE <- yPV
        case 2:  w = ES;   s = -1.f; break;   // iE <- yS
        case 4:  w = PVE;  s =  1.f; break;
        case 5:  w = PVPV; s = -1.f; break;
        case 6:  w = PVS;  s = -1.f; break;
        case 8:  w = SE;   s =  1.f; break;
        case 9:  w = SPV;  s = -1.f; break;
        case 11: w = SV;   s = -1.f; break;   // iS <- yV
        case 12: w = VE;   s =  1.f; break;
        case 14: w = VS;   s = -1.f; break;
        default: return 0.f;
    }
    return s * softplusf(w[lo * din + li]);
}

__global__ __launch_bounds__(256) void prep_wbig_kernel(
    const float* EE, const float* PVE, const float* SE, const float* VE,
    const float* EPV, const float* PVPV, const float* SPV,
    const float* ES, const float* PVS, const float* VS, const float* SV,
    _Float16* __restrict__ Wswz) {
    int t = blockIdx.x * 256 + threadIdx.x;
    const int TOT = KCH * NTILES * 32 * 16;
    if (t >= TOT) return;
    int i = t & 15; int r = t >> 4;
    int l = r & 31; r >>= 5;
    int j = r % NTILES; int c = r / NTILES;
    int n = j * 16 + (l & 15);
    int k = c * 32 + ((l >> 4) << 4) + i;
    Wswz[t] = (_Float16)wbig_val(n, k, EE, PVE, SE, VE, EPV, PVPV, SPV, ES, PVS, VS, SV);
}

// ---------------- prep: input-projection weights (transposed, padded, swizzled) ----------------
__global__ __launch_bounds__(256) void prep_win_kernel(
    const float* wE, const float* wPV, const float* wS, const float* wV,
    _Float16* __restrict__ WinSwz) {
    int t = blockIdx.x * 256 + threadIdx.x;
    const int TOT = KCH_IN * NTILES * 32 * 16;
    if (t >= TOT) return;
    int i = t & 15; int r = t >> 4;
    int l = r & 31; r >>= 5;
    int j = r % NTILES; int c = r / NTILES;
    int n = j * 16 + (l & 15);
    int k = c * 32 + ((l >> 4) << 4) + i;   // k < 768
    float v = 0.f;
    if (n < P_PV)      { int lo = n;        if (lo < D_E)  v = wE [lo * DM + k]; }
    else if (n < P_S)  { int lo = n - P_PV; if (lo < D_PV) v = wPV[lo * DM + k]; }
    else if (n < P_V)  { int lo = n - P_S;  if (lo < D_S)  v = wS [lo * DM + k]; }
    else               { int lo = n - P_V;  if (lo < D_V)  v = wV [lo * DM + k]; }
    WinSwz[t] = (_Float16)v;
}

// ---------------- prep: output weights (transposed, swizzled) ----------------
__global__ __launch_bounds__(256) void prep_wout_kernel(const float* out_w,
                                                        _Float16* __restrict__ WoutSwz) {
    int t = blockIdx.x * 256 + threadIdx.x;
    const int TOT = KCH_IN * NT_OUT * 32 * 16;
    if (t >= TOT) return;
    int i = t & 15; int r = t >> 4;
    int l = r & 31; r >>= 5;
    int j = r % NT_OUT; int c = r / NT_OUT;
    int n = j * 16 + (l & 15);
    int k = c * 32 + ((l >> 4) << 4) + i;
    WoutSwz[t] = (_Float16)out_w[n * DM + k];  // B[k][n] = out_w[n][k]
}

// ---------------- prep: padded input bias + combined recurrent bias ----------------
__global__ __launch_bounds__(256) void prep_bias_kernel(
    const float* bE, const float* bPV, const float* bS, const float* bV,
    const float* bEE, const float* bPVE, const float* bSE, const float* bVE,
    const float* bEPV, const float* bPVPV, const float* bSPV,
    const float* bES, const float* bPVS, const float* bVS, const float* bSV,
    float* __restrict__ bias_in, float* __restrict__ rbias) {
    int n = blockIdx.x * 256 + threadIdx.x;
    if (n >= DPAD) return;
    float bi = 0.f, rb = 0.f;
    if (n < P_PV)      { int lo = n;        if (lo < D_E)  { bi = bE[lo];  rb = bEE[lo] + bEPV[lo] + bES[lo]; } }
    else if (n < P_S)  { int lo = n - P_PV; if (lo < D_PV) { bi = bPV[lo]; rb = bPVE[lo] + bPVPV[lo] + bPVS[lo]; } }
    else if (n < P_V)  { int lo = n - P_S;  if (lo < D_S)  { bi = bS[lo];  rb = bSE[lo] + bSPV[lo] + bSV[lo]; } }
    else               { int lo = n - P_V;  if (lo < D_V)  { bi = bV[lo];  rb = bVE[lo] + bVS[lo]; } }
    bias_in[n] = bi;
    rbias[n] = rb;
}

// ---------------- input projection GEMM: P[t][b][n] = h @ Win^T + bias ----------------
// A-tile (16 contiguous rows of h16) staged once into LDS via TDM, reused by all 8 waves.
__global__ __launch_bounds__(256) void inp_gemm_kernel(const _Float16* __restrict__ h16,
                                                       const _Float16* __restrict__ WinSwz,
                                                       const float* __restrict__ bias_in,
                                                       float* __restrict__ P) {
    __shared__ _Float16 Atile[16 * DM];   // 24.5 KB
    int wave = threadIdx.x >> 5, lane = threadIdx.x & 31;
    int mrow = lane & 15, hi = lane >> 4, kb = hi << 3;
    size_t r0 = (size_t)blockIdx.x * 16;

#ifdef HAVE_TDM
    if (threadIdx.x < 32) {
        tdm_load_1d((unsigned)(size_t)(void*)Atile, h16 + r0 * DM, 16 * DM * (unsigned)sizeof(_Float16));
        __builtin_amdgcn_s_wait_tensorcnt(0);
    }
#else
    for (int i = threadIdx.x; i < 16 * DM; i += 256) Atile[i] = h16[r0 * DM + i];
#endif
    __syncthreads();

    const int ntile = (wave < 2) ? 7 : 6;     // 50 tiles over 8 waves
    int jt[7];
    v8f acc[7];
#pragma unroll
    for (int q = 0; q < 7; ++q) {
        int j = wave + q * 8;
        jt[q] = (j < NTILES) ? j : 0;
        float bv = bias_in[jt[q] * 16 + mrow];
#pragma unroll
        for (int v = 0; v < 8; ++v) acc[q][v] = bv;
    }
    for (int c = 0; c < KCH_IN; ++c) {
        v16h a = load_a_row(&Atile[mrow * DM + c * 32], kb);   // shared across all N tiles
#pragma unroll
        for (int q = 0; q < 7; ++q) {
            if (q >= ntile) continue;
            v16h b = load_b_swz(WinSwz + ((size_t)(c * NTILES + jt[q])) * 512);
            acc[q] = __builtin_amdgcn_wmma_f32_16x16x32_f16(false, a, false, b, (short)0, acc[q], false, false);
        }
    }
#pragma unroll
    for (int q = 0; q < 7; ++q) {
        if (q >= ntile) continue;
#pragma unroll
        for (int v = 0; v < 8; ++v) {
            size_t r = r0 + v + hi * 8;          // global row = b*T + t
            size_t b = r >> 10, t = r & 1023;
            P[(t * BATCH + b) * DPAD + jt[q] * 16 + mrow] = acc[q][v];
        }
    }
}

// ---------------- persistent SSN scan: single workgroup, 16 waves ----------------
__global__ __launch_bounds__(512) void ssn_scan_kernel(const float* __restrict__ P,
                                                       const _Float16* __restrict__ Wswz,
                                                       const float* __restrict__ rbias,
                                                       const int* __restrict__ mask,
                                                       _Float16* __restrict__ Yout) {
    __shared__ _Float16 yb[2][BATCH][DPAD];   // double-buffered activations, 50KB
    const int wave = threadIdx.x >> 5, lane = threadIdx.x & 31;
    const int mrow = lane & 15, hi = lane >> 4, kb = hi << 3;

    int  tiles[4]; bool act[4], isSom[4];
    float rb[4], rate[4];
    v8f x[4], p[4], carry[4];
#pragma unroll
    for (int q = 0; q < 4; ++q) {
        int j = wave + 16 * q;
        tiles[q] = j;
        act[q] = (j < NTILES);
        int jj = act[q] ? j : 0;
        rb[q] = rbias[jj * 16 + mrow];
        float rt;                               // dt / tau per population
        if (jj < 29) rt = 0.3f;                 // E
        else if (jj < 38) rt = 0.6f;            // PV
        else if (jj < 45) rt = 0.1f;            // SOM
        else rt = 0.375f;                       // VIP
        rate[q] = rt;
        isSom[q] = (jj >= 38 && jj < 45);
#pragma unroll
        for (int v = 0; v < 8; ++v) carry[q][v] = 0.f;
    }

    for (int t = 0; t < TLEN; ++t) {
        if (t + 1 < TLEN)
            __builtin_prefetch(&P[((size_t)(t + 1) * BATCH) * DPAD + threadIdx.x * 4], 0, 0);
        // load per-timestep drive, init state
#pragma unroll
        for (int q = 0; q < 4; ++q) {
            if (!act[q]) continue;
            int n = tiles[q] * 16 + mrow;
#pragma unroll
            for (int v = 0; v < 8; ++v) {
                int b = v + hi * 8;
                float pv = P[((size_t)t * BATCH + b) * DPAD + n];
                p[q][v] = pv;
                x[q][v] = isSom[q] ? (0.85f * carry[q][v] + 0.15f * pv) : pv;
            }
        }
        int buf = 0;
        for (int k = 0; k < 6; ++k) {
            // y = 0.04 * relu(x)^2  -> LDS (A-matrix source)
#pragma unroll
            for (int q = 0; q < 4; ++q) {
                if (!act[q]) continue;
                int n = tiles[q] * 16 + mrow;
#pragma unroll
                for (int v = 0; v < 8; ++v) {
                    float r = fmaxf(x[q][v], 0.f);
                    yb[buf][v + hi * 8][n] = (_Float16)(0.04f * r * r);
                }
            }
            __syncthreads();
            // I = Y @ Wbig + p + rbias ; x += rate*(-x + I)
#pragma unroll
            for (int q = 0; q < 4; ++q) {
                if (!act[q]) continue;
                int j = tiles[q];
                v8f acc;
#pragma unroll
                for (int v = 0; v < 8; ++v) acc[v] = p[q][v] + rb[q];
                for (int c = 0; c < KCH; ++c) {
                    v16h a = load_a_row(&yb[buf][mrow][c * 32], kb);
                    v16h b = load_b_swz(Wswz + ((size_t)(c * NTILES + j)) * 512);
                    acc = __builtin_amdgcn_wmma_f32_16x16x32_f16(false, a, false, b, (short)0, acc, false, false);
                }
#pragma unroll
                for (int v = 0; v < 8; ++v) x[q][v] += rate[q] * (acc[v] - x[q][v]);
            }
            __syncthreads();
            buf ^= 1;
        }
        // emit final activations (packed concat layout) + update SOM carry
#pragma unroll
        for (int q = 0; q < 4; ++q) {
            if (!act[q]) continue;
            int n = tiles[q] * 16 + mrow;
            int pk = packed_index(n);
#pragma unroll
            for (int v = 0; v < 8; ++v) {
                int b = v + hi * 8;
                float r = fmaxf(x[q][v], 0.f);
                if (pk >= 0)
                    Yout[((size_t)b * TLEN + t) * DM + pk] = (_Float16)(0.04f * r * r);
                if (isSom[q]) {
                    float m = (float)mask[b * TLEN + t];
                    float nc = 0.85f * carry[q][v] + 0.15f * x[q][v];
                    carry[q][v] = nc * m + carry[q][v] * (1.f - m);
                }
            }
        }
    }
}

// ---------------- output GEMM + residual + LayerNorm (fused, stats in registers) ----------------
__global__ __launch_bounds__(256) void out_gemm_ln_kernel(const _Float16* __restrict__ Y16,
                                                          const _Float16* __restrict__ WoutSwz,
                                                          const float* __restrict__ out_b,
                                                          const float* __restrict__ ln_g,
                                                          const float* __restrict__ ln_b,
                                                          const float* __restrict__ h,
                                                          float* __restrict__ out) {
    __shared__ _Float16 Atile[16 * DM];   // 24.5 KB staged A rows
    __shared__ float red[8][16][2];       // per-wave row stats
    __shared__ float stats[16][2];        // final mu, inv per row
    int wave = threadIdx.x >> 5, lane = threadIdx.x & 31;
    int mrow = lane & 15, hi = lane >> 4, kb = hi << 3;
    size_t r0 = (size_t)blockIdx.x * 16;

#ifdef HAVE_TDM
    if (threadIdx.x < 32) {
        tdm_load_1d((unsigned)(size_t)(void*)Atile, Y16 + r0 * DM, 16 * DM * (unsigned)sizeof(_Float16));
        __builtin_amdgcn_s_wait_tensorcnt(0);
    }
#else
    for (int i = threadIdx.x; i < 16 * DM; i += 256) Atile[i] = Y16[r0 * DM + i];
#endif
    __syncthreads();

    int jt[6];
    v8f acc[6];
#pragma unroll
    for (int q = 0; q < 6; ++q) {
        jt[q] = wave + q * 8;                  // 48 tiles over 8 waves
        float bv = out_b[jt[q] * 16 + mrow];
#pragma unroll
        for (int v = 0; v < 8; ++v) acc[q][v] = bv;
    }
    for (int c = 0; c < KCH_IN; ++c) {
        v16h a = load_a_row(&Atile[mrow * DM + c * 32], kb);
#pragma unroll
        for (int q = 0; q < 6; ++q) {
            v16h b = load_b_swz(WoutSwz + ((size_t)(c * NT_OUT + jt[q])) * 512);
            acc[q] = __builtin_amdgcn_wmma_f32_16x16x32_f16(false, a, false, b, (short)0, acc[q], false, false);
        }
    }
    // residual add + per-lane partial LN stats
    float s[8], s2[8];
#pragma unroll
    for (int v = 0; v < 8; ++v) { s[v] = 0.f; s2[v] = 0.f; }
#pragma unroll
    for (int q = 0; q < 6; ++q) {
        int n = jt[q] * 16 + mrow;
#pragma unroll
        for (int v = 0; v < 8; ++v) {
            int m = v + hi * 8;
            float val = acc[q][v] + h[(r0 + m) * DM + n];
            acc[q][v] = val;
            s[v] += val; s2[v] += val * val;
        }
    }
    // reduce across the 16 lanes of each wave-half (rows 0-7 in lanes 0-15, rows 8-15 in 16-31)
#pragma unroll
    for (int o = 8; o > 0; o >>= 1) {
#pragma unroll
        for (int v = 0; v < 8; ++v) {
            s[v]  += __shfl_xor(s[v],  o, 32);
            s2[v] += __shfl_xor(s2[v], o, 32);
        }
    }
    if (mrow == 0) {   // lanes 0 and 16 publish their half's rows
#pragma unroll
        for (int v = 0; v < 8; ++v) {
            red[wave][v + hi * 8][0] = s[v];
            red[wave][v + hi * 8][1] = s2[v];
        }
    }
    __syncthreads();
    if (threadIdx.x < 16) {
        float ts = 0.f, ts2 = 0.f;
#pragma unroll
        for (int w = 0; w < 8; ++w) { ts += red[w][threadIdx.x][0]; ts2 += red[w][threadIdx.x][1]; }
        float mu = ts * (1.f / DM);
        float var = ts2 * (1.f / DM) - mu * mu;
        stats[threadIdx.x][0] = mu;
        stats[threadIdx.x][1] = rsqrtf(var + 1e-5f);
    }
    __syncthreads();
#pragma unroll
    for (int q = 0; q < 6; ++q) {
        int n = jt[q] * 16 + mrow;
        float g = ln_g[n], bb = ln_b[n];
#pragma unroll
        for (int v = 0; v < 8; ++v) {
            int m = v + hi * 8;
            out[(r0 + m) * DM + n] = (acc[q][v] - stats[m][0]) * stats[m][1] * g + bb;
        }
    }
}

// ---------------- host ----------------
static inline size_t align256(size_t x) { return (x + 255) & ~(size_t)255; }

extern "C" void kernel_launch(void* const* d_in, const int* in_sizes, int n_in,
                              void* d_out, int out_size, void* d_ws, size_t ws_size,
                              hipStream_t stream) {
    (void)in_sizes; (void)n_in; (void)out_size; (void)ws_size;
    const float* h        = (const float*)d_in[0];
    const int*   attn     = (const int*)d_in[1];
    const float* inpE_w   = (const float*)d_in[2];
    const float* inpE_b   = (const float*)d_in[3];
    const float* inpPV_w  = (const float*)d_in[4];
    const float* inpPV_b  = (const float*)d_in[5];
    const float* inpS_w   = (const float*)d_in[6];
    const float* inpS_b   = (const float*)d_in[7];
    const float* inpV_w   = (const float*)d_in[8];
    const float* inpV_b   = (const float*)d_in[9];
    const float* EE_raw   = (const float*)d_in[10]; const float* EE_b   = (const float*)d_in[11];
    const float* PVE_raw  = (const float*)d_in[12]; const float* PVE_b  = (const float*)d_in[13];
    const float* SE_raw   = (const float*)d_in[14]; const float* SE_b   = (const float*)d_in[15];
    const float* VE_raw   = (const float*)d_in[16]; const float* VE_b   = (const float*)d_in[17];
    const float* EPV_raw  = (const float*)d_in[18]; const float* EPV_b  = (const float*)d_in[19];
    const float* PVPV_raw = (const float*)d_in[20]; const float* PVPV_b = (const float*)d_in[21];
    const float* SPV_raw  = (const float*)d_in[22]; const float* SPV_b  = (const float*)d_in[23];
    const float* ES_raw   = (const float*)d_in[24]; const float* ES_b   = (const float*)d_in[25];
    const float* PVS_raw  = (const float*)d_in[26]; const float* PVS_b  = (const float*)d_in[27];
    const float* VS_raw   = (const float*)d_in[28]; const float* VS_b   = (const float*)d_in[29];
    const float* SV_raw   = (const float*)d_in[30]; const float* SV_b   = (const float*)d_in[31];
    const float* out_w    = (const float*)d_in[32];
    const float* out_b    = (const float*)d_in[33];
    const float* ln_g     = (const float*)d_in[34];
    const float* ln_b     = (const float*)d_in[35];
    float* out = (float*)d_out;

    // workspace carve-out
    char* ws = (char*)d_ws;
    size_t off = 0;
    float*     P        = (float*)(ws + off);     off += align256((size_t)TLEN * BATCH * DPAD * 4);
    _Float16*  h16      = (_Float16*)(ws + off);  off += align256((size_t)BATCH * TLEN * DM * 2);
    _Float16*  Y16      = (_Float16*)(ws + off);  off += align256((size_t)BATCH * TLEN * DM * 2);
    _Float16*  Wswz     = (_Float16*)(ws + off);  off += align256((size_t)KCH * NTILES * 512 * 2);
    _Float16*  WinSwz   = (_Float16*)(ws + off);  off += align256((size_t)KCH_IN * NTILES * 512 * 2);
    _Float16*  WoutSwz  = (_Float16*)(ws + off);  off += align256((size_t)KCH_IN * NT_OUT * 512 * 2);
    float*     bias_in  = (float*)(ws + off);     off += align256(DPAD * 4);
    float*     rbias    = (float*)(ws + off);     off += align256(DPAD * 4);

    const int nH = BATCH * TLEN * DM;
    h_to_f16_kernel<<<(nH + 255) / 256, 256, 0, stream>>>(h, h16, nH);
    prep_wbig_kernel<<<(KCH * NTILES * 512 + 255) / 256, 256, 0, stream>>>(
        EE_raw, PVE_raw, SE_raw, VE_raw, EPV_raw, PVPV_raw, SPV_raw,
        ES_raw, PVS_raw, VS_raw, SV_raw, Wswz);
    prep_win_kernel<<<(KCH_IN * NTILES * 512 + 255) / 256, 256, 0, stream>>>(
        inpE_w, inpPV_w, inpS_w, inpV_w, WinSwz);
    prep_wout_kernel<<<(KCH_IN * NT_OUT * 512 + 255) / 256, 256, 0, stream>>>(out_w, WoutSwz);
    prep_bias_kernel<<<(DPAD + 255) / 256, 256, 0, stream>>>(
        inpE_b, inpPV_b, inpS_b, inpV_b,
        EE_b, PVE_b, SE_b, VE_b, EPV_b, PVPV_b, SPV_b, ES_b, PVS_b, VS_b, SV_b,
        bias_in, rbias);

    inp_gemm_kernel<<<TLEN * BATCH / 16, 256, 0, stream>>>(h16, WinSwz, bias_in, P);
    ssn_scan_kernel<<<1, 512, 0, stream>>>(P, Wswz, rbias, attn, Y16);
    out_gemm_ln_kernel<<<TLEN * BATCH / 16, 256, 0, stream>>>(Y16, WoutSwz, out_b, ln_g, ln_b, h, out);
}
